// ExpertChoiceMoELayer_1726576853054
// MI455X (gfx1250) — matrix operand order repaired and checked by reading, not compiled, not run on hardware
//
#include <hip/hip_runtime.h>

// ---------------------------------------------------------------------------
// Expert-Choice MoE (B=2,S=2048,H=1024,I=4096,E=8, cap=512) for gfx1250.
// bf16 v_wmma_f32_16x16x32_bf16 GEMMs, double-buffered LDS, and CDNA5
// global_load_async_to_lds_b128 DMA for the bf16 activation tiles.
// Weights stay fp32 in HBM (402 MB ~= 17us @ 23.3 TB/s) and are converted
// fp32->bf16 on the fly while staging into LDS.
// ---------------------------------------------------------------------------

#define NTOK 4096
#define HDIM 1024
#define IDIM 4096
#define EEXP 8
#define CAP  512

#define BM 128
#define BN 64
#define BK 32
#define BKP 40            // LDS row stride in bf16 elems (80B) -> conflict-free b128 reads
#define TA (BM * BKP)     // A tile elems per buffer
#define TB (BN * BKP)     // B tile elems per buffer

typedef __attribute__((ext_vector_type(16))) __bf16 v16bf;
typedef __attribute__((ext_vector_type(2)))  __bf16 v2bf;
typedef __attribute__((ext_vector_type(8)))  float  v8f;

#if __has_builtin(__builtin_amdgcn_global_load_async_to_lds_b128) && \
    __has_builtin(__builtin_amdgcn_s_wait_asynccnt)
#define USE_ASYNC_COPY 1
typedef int v4i __attribute__((vector_size(16)));            // int4 GCC vector
typedef __attribute__((address_space(1))) v4i* g128_t;       // global int4*
typedef __attribute__((address_space(3))) v4i* l128_t;       // LDS int4*
static __device__ inline void async_b128(const __bf16* g, __bf16* l) {
  __builtin_amdgcn_global_load_async_to_lds_b128(
      (g128_t)(v4i*)(uintptr_t)g,  // generic->AS1 (identity for global VAs)
      (l128_t)(v4i*)l,             // generic->AS3 addrspacecast
      0, 0);
}
static __device__ inline void async_wait0() {
  __builtin_amdgcn_s_wait_asynccnt(0);
}
#else
#define USE_ASYNC_COPY 0
static __device__ inline void async_wait0() {}
#endif

static __device__ inline unsigned int pack2(float a, float b) {
  union { v2bf v; unsigned int u; } q;
  q.v.x = (__bf16)a;
  q.v.y = (__bf16)b;
  return q.u;
}

static __device__ inline uint4 cvt8(float4 a, float4 b) {
  uint4 r = { pack2(a.x, a.y), pack2(a.z, a.w), pack2(b.x, b.y), pack2(b.z, b.w) };
  return r;
}

static __device__ inline v16bf load_frag(const __bf16* p0, const __bf16* p1) {
  union { v16bf v; uint4 q[2]; } u;
  u.q[0] = *(const uint4*)p0;
  u.q[1] = *(const uint4*)p1;
  return u.v;
}

static __device__ inline v8f wmma_bf16(v16bf a, v16bf b, v8f c) {
  return __builtin_amdgcn_wmma_f32_16x16x32_bf16(false, a, false, b, (short)0, c,
                                                 false, false);
}

// ---------------------------------------------------------------------------
// 0) zero scratch (accum NH + counts N + aux 1, contiguous)
// ---------------------------------------------------------------------------
__global__ __launch_bounds__(256) void zero_kernel(float* p, int n) {
  for (int i = blockIdx.x * 256 + threadIdx.x; i < n; i += gridDim.x * 256)
    p[i] = 0.f;
}

// ---------------------------------------------------------------------------
// 1) gating: logits[N,E] = x @ w_gate^T ; aux += logsumexp(logits_row)^2
// ---------------------------------------------------------------------------
__global__ __launch_bounds__(256) void gating_kernel(
    const float* __restrict__ x, const float* __restrict__ w_gate,
    float* __restrict__ logits, float* __restrict__ aux) {
  const int t = blockIdx.x, tid = threadIdx.x;
  float part[EEXP];
#pragma unroll
  for (int e = 0; e < EEXP; ++e) part[e] = 0.f;
  for (int h = tid; h < HDIM; h += 256) {
    float xv = x[(size_t)t * HDIM + h];
#pragma unroll
    for (int e = 0; e < EEXP; ++e) part[e] += xv * w_gate[e * HDIM + h];
  }
  __shared__ float red[256];
  __shared__ float lg[EEXP];
  for (int e = 0; e < EEXP; ++e) {
    red[tid] = part[e];
    __syncthreads();
    for (int s = 128; s > 0; s >>= 1) {
      if (tid < s) red[tid] += red[tid + s];
      __syncthreads();
    }
    if (tid == 0) lg[e] = red[0];
    __syncthreads();
  }
  if (tid == 0) {
    float m = -3.402823466e38f;
#pragma unroll
    for (int e = 0; e < EEXP; ++e) m = fmaxf(m, lg[e]);
    float s = 0.f;
#pragma unroll
    for (int e = 0; e < EEXP; ++e) s += __expf(lg[e] - m);
    float lse = m + __logf(s);
    atomicAdd(aux, lse * lse);
#pragma unroll
    for (int e = 0; e < EEXP; ++e) logits[t * EEXP + e] = lg[e];
  }
}

// ---------------------------------------------------------------------------
// 2) per-expert column softmax stats (softmax over token axis)
// ---------------------------------------------------------------------------
__global__ __launch_bounds__(256) void stats_kernel(
    const float* __restrict__ logits, float* __restrict__ maxv,
    float* __restrict__ sumv) {
  const int e = blockIdx.x, tid = threadIdx.x;
  __shared__ float red[256];
  float m = -3.402823466e38f;
  for (int n = tid; n < NTOK; n += 256) m = fmaxf(m, logits[n * EEXP + e]);
  red[tid] = m;
  __syncthreads();
  for (int s = 128; s > 0; s >>= 1) {
    if (tid < s) red[tid] = fmaxf(red[tid], red[tid + s]);
    __syncthreads();
  }
  m = red[0];
  __syncthreads();
  float sum = 0.f;
  for (int n = tid; n < NTOK; n += 256) sum += __expf(logits[n * EEXP + e] - m);
  red[tid] = sum;
  __syncthreads();
  for (int s = 128; s > 0; s >>= 1) {
    if (tid < s) red[tid] += red[tid + s];
    __syncthreads();
  }
  if (tid == 0) { maxv[e] = m; sumv[e] = red[0]; }
}

// ---------------------------------------------------------------------------
// 3) per-expert top-CAP via radix select (monotonic key; ties -> lowest index)
// ---------------------------------------------------------------------------
static __device__ inline unsigned key_of(const float* logits, int n, int e) {
  unsigned u = __float_as_uint(logits[n * EEXP + e]);
  return (u & 0x80000000u) ? ~u : (u | 0x80000000u);
}

__global__ __launch_bounds__(256) void topk_kernel(
    const float* __restrict__ logits, const float* __restrict__ maxv,
    const float* __restrict__ sumv, int* __restrict__ top_i,
    float* __restrict__ top_p, float* __restrict__ counts) {
  const int e = blockIdx.x, tid = threadIdx.x;
  __shared__ unsigned s_cnt;
  __shared__ unsigned s_pos;
  unsigned prefix = 0, remaining = CAP;
  for (int bit = 31; bit >= 0; --bit) {
    if (tid == 0) s_cnt = 0;
    __syncthreads();
    const unsigned msk = 0xFFFFFFFFu << bit;
    const unsigned want = prefix | (1u << bit);
    unsigned local = 0;
    for (int n = tid; n < NTOK; n += 256)
      if ((key_of(logits, n, e) & msk) == want) local++;
    atomicAdd(&s_cnt, local);
    __syncthreads();
    unsigned c = s_cnt;
    __syncthreads();
    if (c >= remaining) prefix = want; else remaining -= c;
  }
  // prefix == CAP-th largest key
  if (tid == 0) s_pos = 0;
  __syncthreads();
  const float mx = maxv[e], sm = sumv[e];
  for (int n = tid; n < NTOK; n += 256) {
    if (key_of(logits, n, e) > prefix) {
      unsigned pos = atomicAdd(&s_pos, 1u);
      float p = __expf(logits[n * EEXP + e] - mx) / sm;
      top_i[e * CAP + pos] = n;
      top_p[e * CAP + pos] = p;
      atomicAdd(&counts[n], p);
    }
  }
  __syncthreads();
  if (tid == 0) {  // stable tie fill, lowest index first (rare path)
    unsigned pos = s_pos;
    for (int n = 0; n < NTOK && pos < CAP; ++n) {
      if (key_of(logits, n, e) == prefix) {
        float p = __expf(logits[n * EEXP + e] - mx) / sm;
        top_i[e * CAP + pos] = n;
        top_p[e * CAP + pos] = p;
        atomicAdd(&counts[n], p);
        pos++;
      }
    }
  }
}

// ---------------------------------------------------------------------------
// 4) gather selected token rows, fp32 -> bf16
// ---------------------------------------------------------------------------
__global__ __launch_bounds__(256) void gather_kernel(
    const float* __restrict__ x, const int* __restrict__ top_i,
    __bf16* __restrict__ xe) {
  const int idx = blockIdx.x;  // e*CAP + c
  const int tok = top_i[idx];
  const float* src = x + (size_t)tok * HDIM;
  __bf16* dst = xe + (size_t)idx * HDIM;
  const int h = threadIdx.x * 4;
  float4 f = *(const float4*)(src + h);
  uint2 d = { pack2(f.x, f.y), pack2(f.z, f.w) };
  *(uint2*)(dst + h) = d;
}

// ---------------------------------------------------------------------------
// 5) FFN1: act[e] = silu(xe @ wg^T) * (xe @ wu^T)   (bf16 out)
//    128x64 block tile, 8 waves (4x2) of 32x32; double-buffered LDS;
//    A tile via async DMA to LDS (bf16), weight tiles via reg cvt pipeline.
// ---------------------------------------------------------------------------
__global__ __launch_bounds__(256) void ffn1_kernel(
    const __bf16* __restrict__ xe, const float* __restrict__ wg,
    const float* __restrict__ wu, __bf16* __restrict__ act) {
  const int e = blockIdx.z;
  const int m0 = blockIdx.y * BM;
  const int n0 = blockIdx.x * BN;

  __shared__ __bf16 As[2 * TA];
  __shared__ __bf16 Gs[2 * TB];
  __shared__ __bf16 Us[2 * TB];

  const int tid = threadIdx.x;
  const int lane = tid & 31, wave = tid >> 5;
  const int wm = wave & 3, wn = wave >> 2;
  const int half = lane >> 4, lr = lane & 15;

  const int ar = tid >> 1, as_ = (tid & 1) * 16;   // A: 128 rows x 32 bf16
  const int br = tid >> 2, bs = (tid & 3) * 8;     // B: 64 rows x 32 fp32

  const __bf16* a_src = xe + (size_t)e * CAP * HDIM + (size_t)(m0 + ar) * HDIM + as_;
  const float*  g_src = wg + (size_t)e * IDIM * HDIM + (size_t)(n0 + br) * HDIM + bs;
  const float*  u_src = wu + (size_t)e * IDIM * HDIM + (size_t)(n0 + br) * HDIM + bs;

  const v8f vz = {0.f, 0.f, 0.f, 0.f, 0.f, 0.f, 0.f, 0.f};
  v8f accg[2][2] = {{vz, vz}, {vz, vz}};
  v8f accu[2][2] = {{vz, vz}, {vz, vz}};

  // ---- prologue: tile k0=0 into buffer 0
#if USE_ASYNC_COPY
  async_b128(a_src, As + ar * BKP + as_);
  async_b128(a_src + 8, As + ar * BKP + as_ + 8);
#else
  {
    uint4 d0 = *(const uint4*)a_src;
    uint4 d1 = *(const uint4*)(a_src + 8);
    *(uint4*)(As + ar * BKP + as_)     = d0;
    *(uint4*)(As + ar * BKP + as_ + 8) = d1;
  }
#endif
  {
    float4 f0 = *(const float4*)g_src, f1 = *(const float4*)(g_src + 4);
    *(uint4*)(Gs + br * BKP + bs) = cvt8(f0, f1);
    float4 h0 = *(const float4*)u_src, h1 = *(const float4*)(u_src + 4);
    *(uint4*)(Us + br * BKP + bs) = cvt8(h0, h1);
  }
  async_wait0();
  __syncthreads();

  int p = 0;
  for (int k0 = 0; k0 < HDIM; k0 += BK) {
    const bool more = (k0 + BK) < HDIM;
    const int np = p ^ 1;

    // 1) prefetch next tile (A via async DMA straight into LDS buffer np)
    float4 f0, f1, h0, h1;
#if !USE_ASYNC_COPY
    uint4 pa0, pa1;
#endif
    if (more) {
#if USE_ASYNC_COPY
      async_b128(a_src + k0 + BK, As + np * TA + ar * BKP + as_);
      async_b128(a_src + k0 + BK + 8, As + np * TA + ar * BKP + as_ + 8);
#else
      pa0 = *(const uint4*)(a_src + k0 + BK);
      pa1 = *(const uint4*)(a_src + k0 + BK + 8);
#endif
      f0 = *(const float4*)(g_src + k0 + BK);
      f1 = *(const float4*)(g_src + k0 + BK + 4);
      h0 = *(const float4*)(u_src + k0 + BK);
      h1 = *(const float4*)(u_src + k0 + BK + 4);
    }

    // 2) compute on buffer p
    const __bf16* Ab = As + p * TA;
    const __bf16* Gb = Gs + p * TB;
    const __bf16* Ub = Us + p * TB;
    v16bf af[2], gf[2], uf[2];
#pragma unroll
    for (int tm = 0; tm < 2; ++tm) {
      const __bf16* q = Ab + (wm * 32 + tm * 16 + lr) * BKP + half * 16;
      af[tm] = load_frag(q, q + 8);    // A: K-contiguous per row, lane half h holds k 16h..16h+15
    }
#pragma unroll
    for (int tn = 0; tn < 2; ++tn) {
      const __bf16* qg = Gb + (wn * 32 + tn * 16 + lr) * BKP + half * 16;
      gf[tn] = load_frag(qg, qg + 8);  // B: K-contiguous per output column
      const __bf16* qu = Ub + (wn * 32 + tn * 16 + lr) * BKP + half * 16;
      uf[tn] = load_frag(qu, qu + 8);
    }
#pragma unroll
    for (int tm = 0; tm < 2; ++tm)
#pragma unroll
      for (int tn = 0; tn < 2; ++tn) {
        accg[tm][tn] = wmma_bf16(af[tm], gf[tn], accg[tm][tn]);
        accu[tm][tn] = wmma_bf16(af[tm], uf[tn], accu[tm][tn]);
      }

    // 3) commit prefetched tile into buffer np, 4) fence + swap
    if (more) {
#if !USE_ASYNC_COPY
      *(uint4*)(As + np * TA + ar * BKP + as_)     = pa0;
      *(uint4*)(As + np * TA + ar * BKP + as_ + 8) = pa1;
#endif
      *(uint4*)(Gs + np * TB + br * BKP + bs) = cvt8(f0, f1);
      *(uint4*)(Us + np * TB + br * BKP + bs) = cvt8(h0, h1);
      async_wait0();
      __syncthreads();
      p = np;
    }
  }

  // epilogue: act = silu(g) * u  (fast v_rcp_f32 sigmoid)
  __bf16* dst = act + (size_t)e * CAP * IDIM;
#pragma unroll
  for (int tm = 0; tm < 2; ++tm)
#pragma unroll
    for (int tn = 0; tn < 2; ++tn)
#pragma unroll
      for (int v = 0; v < 8; ++v) {
        int row = m0 + wm * 32 + tm * 16 + half * 8 + v;
        int col = n0 + wn * 32 + tn * 16 + lr;
        float g = accg[tm][tn][v];
        float u = accu[tm][tn][v];
        float s = g * __builtin_amdgcn_rcpf(1.f + __expf(-g));
        dst[(size_t)row * IDIM + col] = (__bf16)(s * u);
      }
}

// ---------------------------------------------------------------------------
// 6) FFN2: out = act @ wd^T, scaled by top_p, scatter-add into accum
// ---------------------------------------------------------------------------
__global__ __launch_bounds__(256) void ffn2_kernel(
    const __bf16* __restrict__ act, const float* __restrict__ wd,
    const int* __restrict__ top_i, const float* __restrict__ top_p,
    float* __restrict__ accum) {
  const int e = blockIdx.z;
  const int m0 = blockIdx.y * BM;
  const int n0 = blockIdx.x * BN;

  __shared__ __bf16 As[2 * TA];
  __shared__ __bf16 Bs[2 * TB];

  const int tid = threadIdx.x;
  const int lane = tid & 31, wave = tid >> 5;
  const int wm = wave & 3, wn = wave >> 2;
  const int half = lane >> 4, lr = lane & 15;

  const int ar = tid >> 1, as_ = (tid & 1) * 16;
  const int br = tid >> 2, bs = (tid & 3) * 8;

  const __bf16* a_src = act + (size_t)e * CAP * IDIM + (size_t)(m0 + ar) * IDIM + as_;
  const float*  b_src = wd  + (size_t)e * HDIM * IDIM + (size_t)(n0 + br) * IDIM + bs;

  const v8f vz = {0.f, 0.f, 0.f, 0.f, 0.f, 0.f, 0.f, 0.f};
  v8f acc[2][2] = {{vz, vz}, {vz, vz}};

#if USE_ASYNC_COPY
  async_b128(a_src, As + ar * BKP + as_);
  async_b128(a_src + 8, As + ar * BKP + as_ + 8);
#else
  {
    uint4 d0 = *(const uint4*)a_src;
    uint4 d1 = *(const uint4*)(a_src + 8);
    *(uint4*)(As + ar * BKP + as_)     = d0;
    *(uint4*)(As + ar * BKP + as_ + 8) = d1;
  }
#endif
  {
    float4 f0 = *(const float4*)b_src, f1 = *(const float4*)(b_src + 4);
    *(uint4*)(Bs + br * BKP + bs) = cvt8(f0, f1);
  }
  async_wait0();
  __syncthreads();

  int p = 0;
  for (int k0 = 0; k0 < IDIM; k0 += BK) {
    const bool more = (k0 + BK) < IDIM;
    const int np = p ^ 1;

    float4 f0, f1;
#if !USE_ASYNC_COPY
    uint4 pa0, pa1;
#endif
    if (more) {
#if USE_ASYNC_COPY
      async_b128(a_src + k0 + BK, As + np * TA + ar * BKP + as_);
      async_b128(a_src + k0 + BK + 8, As + np * TA + ar * BKP + as_ + 8);
#else
      pa0 = *(const uint4*)(a_src + k0 + BK);
      pa1 = *(const uint4*)(a_src + k0 + BK + 8);
#endif
      f0 = *(const float4*)(b_src + k0 + BK);
      f1 = *(const float4*)(b_src + k0 + BK + 4);
    }

    const __bf16* Ab = As + p * TA;
    const __bf16* Bb = Bs + p * TB;
    v16bf af[2], bf[2];
#pragma unroll
    for (int tm = 0; tm < 2; ++tm) {
      const __bf16* q = Ab + (wm * 32 + tm * 16 + lr) * BKP + half * 16;
      af[tm] = load_frag(q, q + 8);
    }
#pragma unroll
    for (int tn = 0; tn < 2; ++tn) {
      const __bf16* q = Bb + (wn * 32 + tn * 16 + lr) * BKP + half * 16;
      bf[tn] = load_frag(q, q + 8);
    }
#pragma unroll
    for (int tm = 0; tm < 2; ++tm)
#pragma unroll
      for (int tn = 0; tn < 2; ++tn)
        acc[tm][tn] = wmma_bf16(af[tm], bf[tn], acc[tm][tn]);

    if (more) {
#if !USE_ASYNC_COPY
      *(uint4*)(As + np * TA + ar * BKP + as_)     = pa0;
      *(uint4*)(As + np * TA + ar * BKP + as_ + 8) = pa1;
#endif
      *(uint4*)(Bs + np * TB + br * BKP + bs) = cvt8(f0, f1);
      async_wait0();
      __syncthreads();
      p = np;
    }
  }

#pragma unroll
  for (int tm = 0; tm < 2; ++tm)
#pragma unroll
    for (int tn = 0; tn < 2; ++tn)
#pragma unroll
      for (int v = 0; v < 8; ++v) {
        int c = m0 + wm * 32 + tm * 16 + half * 8 + v;
        int h = n0 + wn * 32 + tn * 16 + lr;
        int tok = top_i[e * CAP + c];
        float p2 = top_p[e * CAP + c];
        atomicAdd(&accum[(size_t)tok * HDIM + h], p2 * acc[tm][tn][v]);
      }
}

// ---------------------------------------------------------------------------
// 7) finalize: out = accum / clip(counts,1e-9); aux at tail
// ---------------------------------------------------------------------------
__global__ __launch_bounds__(256) void finalize_kernel(
    const float* __restrict__ accum, const float* __restrict__ counts,
    const float* __restrict__ aux, float* __restrict__ out) {
  const unsigned gid = blockIdx.x * 256u + threadIdx.x;
  const unsigned NH = (unsigned)NTOK * HDIM;
  if (gid < NH) {
    const unsigned n = gid >> 10;  // /HDIM
    float c = fmaxf(counts[n], 1e-9f);
    out[gid] = accum[gid] / c;
  }
  if (gid == 0) out[NH] = 0.001f * (*aux) / (float)NTOK;
}

// ---------------------------------------------------------------------------
extern "C" void kernel_launch(void* const* d_in, const int* in_sizes, int n_in,
                              void* d_out, int out_size, void* d_ws,
                              size_t ws_size, hipStream_t stream) {
  const float* x      = (const float*)d_in[0];
  const float* w_gate = (const float*)d_in[1];
  const float* wg     = (const float*)d_in[2];
  const float* wu     = (const float*)d_in[3];
  const float* wd     = (const float*)d_in[4];
  float* out = (float*)d_out;

  char* base = (char*)d_ws;
  size_t off = 0;
  auto alloc = [&](size_t bytes) -> void* {
    void* p = base + off;
    off = (off + bytes + 255) & ~(size_t)255;
    return p;
  };

  const size_t NH = (size_t)NTOK * HDIM;
  // accum + counts + aux contiguous so one zero pass covers them
  float* accum  = (float*)alloc((NH + NTOK + 1) * sizeof(float));
  float* counts = accum + NH;
  float* aux    = counts + NTOK;
  float* logits = (float*)alloc((size_t)NTOK * EEXP * sizeof(float));
  float* maxv   = (float*)alloc(EEXP * sizeof(float));
  float* sumv   = (float*)alloc(EEXP * sizeof(float));
  int*   top_i  = (int*)  alloc((size_t)EEXP * CAP * sizeof(int));
  float* top_p  = (float*)alloc((size_t)EEXP * CAP * sizeof(float));
  __bf16* xe    = (__bf16*)alloc((size_t)EEXP * CAP * HDIM * sizeof(__bf16));
  __bf16* act   = (__bf16*)alloc((size_t)EEXP * CAP * IDIM * sizeof(__bf16));
  (void)ws_size;  // requires ~59 MB of scratch

  zero_kernel<<<2048, 256, 0, stream>>>(accum, (int)(NH + NTOK + 1));
  gating_kernel<<<NTOK, 256, 0, stream>>>(x, w_gate, logits, aux);
  stats_kernel<<<EEXP, 256, 0, stream>>>(logits, maxv, sumv);
  topk_kernel<<<EEXP, 256, 0, stream>>>(logits, maxv, sumv, top_i, top_p, counts);
  gather_kernel<<<EEXP * CAP, 256, 0, stream>>>(x, top_i, xe);
  dim3 g1(IDIM / BN, CAP / BM, EEXP);   // (64,4,8)
  ffn1_kernel<<<g1, 256, 0, stream>>>(xe, wg, wu, act);
  dim3 g2(HDIM / BN, CAP / BM, EEXP);   // (16,4,8)
  ffn2_kernel<<<g2, 256, 0, stream>>>(act, wd, top_i, top_p, accum);
  finalize_kernel<<<(unsigned)(NH / 256), 256, 0, stream>>>(accum, counts, aux, out);
}
